// MeshDiT_35347580846134
// MI455X (gfx1250) — 8-bit matrix operand order repaired and checked by reading, compiled, not run on hardware
//
#include <hip/hip_runtime.h>
#include <hip/hip_bf16.h>
#include <cstdint>
#include <cstddef>


#define HDIM   768
#define HEADS  12
#define INTER  3072
#define BATCH  2
#define TXX    1024
#define TCC    256
#define TT     (TXX + TCC)

typedef __attribute__((ext_vector_type(8))) int v8i;

__device__ __forceinline__ float sigm_(float x){ return 1.f/(1.f+__expf(-x)); }
__device__ __forceinline__ float silu_(float x){ return x*sigm_(x); }

__global__ void k_timestep(const float* __restrict__ t, float* __restrict__ emb){
  int idx = blockIdx.x*blockDim.x + threadIdx.x;
  if (idx >= BATCH*256) return;
  int b = idx >> 8, j = idx & 255;
  float fr = __expf(-logf(10000.f) * (float)(j & 127) / 128.f);
  float a  = t[b] * fr;
  emb[idx] = (j < 128) ? cosf(a) : sinf(a);
}

__global__ void k_linear(const float* __restrict__ X, const float* __restrict__ W,
                         const float* __restrict__ bias, float* __restrict__ Y,
                         int M, int N, int K, int presilu){
  int idx = blockIdx.x*blockDim.x + threadIdx.x;
  if (idx >= M*N) return;
  int m = idx / N, n = idx % N;
  const float* xr = X + (size_t)m*K;
  const float* wr = W + (size_t)n*K;
  float acc = bias[n];
  for (int k = 0; k < K; ++k){
    float v = xr[k];
    if (presilu) v = silu_(v);
    acc += v * wr[k];
  }
  Y[idx] = acc;
}

__global__ void k_wq_partial(const float* __restrict__ W, size_t n, float* __restrict__ part){
  __shared__ float red[256];
  float s = 0.f;
  for (size_t i = (size_t)blockIdx.x*256 + threadIdx.x; i < n; i += 256*256)
    s += fabsf(W[i]);
  red[threadIdx.x] = s; __syncthreads();
  for (int o = 128; o > 0; o >>= 1){
    if (threadIdx.x < o) red[threadIdx.x] += red[threadIdx.x+o];
    __syncthreads();
  }
  if (threadIdx.x == 0) part[blockIdx.x] = red[0];
}
__global__ void k_wq_fin(const float* __restrict__ part, size_t n, float* __restrict__ slot){
  __shared__ float red[256];
  red[threadIdx.x] = part[threadIdx.x]; __syncthreads();
  for (int o = 128; o > 0; o >>= 1){
    if (threadIdx.x < o) red[threadIdx.x] += red[threadIdx.x+o];
    __syncthreads();
  }
  if (threadIdx.x == 0){
    float m = fmaxf(red[0]/(float)n, 1e-5f);
    slot[0] = 1.f/m;
    slot[1] = m;
  }
}
__global__ void k_wq_quant(const float* __restrict__ W, const float* __restrict__ slot,
                           int8_t* __restrict__ qw, size_t n){
  size_t i = (size_t)blockIdx.x*256 + threadIdx.x;
  if (i >= n) return;
  float q = rintf(W[i]*slot[0]);
  q = fminf(fmaxf(q, -1.f), 1.f);
  qw[i] = (int8_t)q;
}

__global__ void k_rq(const float* __restrict__ X, const float* __restrict__ g,
                     int8_t* __restrict__ qx, float* __restrict__ rowdq, int K){
  __shared__ float sx[3072];
  __shared__ float red[256];
  int row = blockIdx.x;
  const float* xr = X + (size_t)row*K;
  float ss = 0.f;
  for (int k = threadIdx.x; k < K; k += 256){ float v = xr[k]; sx[k] = v; ss += v*v; }
  red[threadIdx.x] = ss; __syncthreads();
  for (int o = 128; o > 0; o >>= 1){
    if (threadIdx.x < o) red[threadIdx.x] += red[threadIdx.x+o];
    __syncthreads();
  }
  float rms = rsqrtf(red[0]/(float)K + 1e-8f);
  __syncthreads();
  float amax = 0.f;
  for (int k = threadIdx.x; k < K; k += 256){
    float v = sx[k]*rms*g[k]; sx[k] = v; amax = fmaxf(amax, fabsf(v));
  }
  red[threadIdx.x] = amax; __syncthreads();
  for (int o = 128; o > 0; o >>= 1){
    if (threadIdx.x < o) red[threadIdx.x] = fmaxf(red[threadIdx.x], red[threadIdx.x+o]);
    __syncthreads();
  }
  float sc = 127.f / fmaxf(red[0], 1e-5f);
  if (threadIdx.x == 0) rowdq[row] = 1.f/sc;
  for (int k = threadIdx.x; k < K; k += 256){
    float q = fminf(fmaxf(rintf(sx[k]*sc), -128.f), 127.f);
    qx[(size_t)row*K + k] = (int8_t)q;
  }
}

__global__ void __launch_bounds__(256)
k_gemm_i8(const int8_t* __restrict__ qx, const int8_t* __restrict__ qw,
          const float* __restrict__ rowdq, const float* __restrict__ wslot,
          const float* __restrict__ bias, float* __restrict__ Y,
          int M, int N, int K, int ldy, int col0){
  __shared__ char As[2*4096];
  const int tid  = threadIdx.x;
  const int wave = tid >> 5, lane = tid & 31;
  const int wm = wave >> 2, wn = wave & 3;
  const int l16 = lane & 15, hi = lane >> 4;
  const int tMblk = blockIdx.y * 64;
  const int tN    = blockIdx.x * 128 + wn * 32;

  const int srow = tid >> 2, sseg = tid & 3;
  const uint64_t gA = (uint64_t)(uintptr_t)(qx + (size_t)(tMblk + srow)*K + sseg*16);
  const int nk = K >> 6;

  #define ISSUE(chunk) do {                                                   \
    unsigned _ldso = (unsigned)(((unsigned)((chunk) & 1) << 12) + srow*64 + sseg*16); \
    uint64_t _ga   = gA + ((uint64_t)(chunk) << 6);                           \
    asm volatile("global_load_async_to_lds_b128 %0, %1, off"                  \
                 :: "v"(_ldso), "v"(_ga) : "memory");                         \
  } while (0)

  ISSUE(0);
  if (nk > 1) ISSUE(1);

  v8i acc00 = {}, acc01 = {}, acc10 = {}, acc11 = {};
  const int8_t* bx0 = qw + (size_t)(tN + l16)*K + hi*16;
  const int8_t* bx1 = qw + (size_t)(tN + 16 + l16)*K + hi*16;
  const int arow0 = wm*32 + l16;
  const int arow1 = wm*32 + 16 + l16;

  for (int i = 0; i < nk; ++i){
    if (i + 1 < nk) asm volatile("s_wait_asynccnt 0x1" ::: "memory");
    else            asm volatile("s_wait_asynccnt 0x0" ::: "memory");
    __syncthreads();
    const char* Ab = As + ((i & 1) << 12);
    v8i a0, a1;
    {
      const char* p = Ab + arow0*64 + hi*16;
      int2 x0 = *(const int2*)(p);      int2 x1 = *(const int2*)(p + 8);
      int2 x2 = *(const int2*)(p + 32); int2 x3 = *(const int2*)(p + 40);
      a0[0]=x0.x; a0[1]=x0.y; a0[2]=x1.x; a0[3]=x1.y;
      a0[4]=x2.x; a0[5]=x2.y; a0[6]=x3.x; a0[7]=x3.y;
    }
    {
      const char* p = Ab + arow1*64 + hi*16;
      int2 x0 = *(const int2*)(p);      int2 x1 = *(const int2*)(p + 8);
      int2 x2 = *(const int2*)(p + 32); int2 x3 = *(const int2*)(p + 40);
      a1[0]=x0.x; a1[1]=x0.y; a1[2]=x1.x; a1[3]=x1.y;
      a1[4]=x2.x; a1[5]=x2.y; a1[6]=x3.x; a1[7]=x3.y;
    }
    const int k = i << 6;
    if (i + 1 < nk){
      __builtin_prefetch(bx0 + k + 64, 0, 0);
      __builtin_prefetch(bx1 + k + 64, 0, 0);
    }
    int4 p00 = *(const int4*)(bx0 + k);  int4 p01 = *(const int4*)(bx0 + k + 32);
    int4 p10 = *(const int4*)(bx1 + k);  int4 p11 = *(const int4*)(bx1 + k + 32);
    v8i b0; b0[0]=p00.x; b0[1]=p00.y; b0[2]=p00.z; b0[3]=p00.w;
            b0[4]=p01.x; b0[5]=p01.y; b0[6]=p01.z; b0[7]=p01.w;
    v8i b1; b1[0]=p10.x; b1[1]=p10.y; b1[2]=p10.z; b1[3]=p10.w;
            b1[4]=p11.x; b1[5]=p11.y; b1[6]=p11.z; b1[7]=p11.w;
    acc00 = __builtin_amdgcn_wmma_i32_16x16x64_iu8(true, a0, true, b0, acc00, false, false);
    acc01 = __builtin_amdgcn_wmma_i32_16x16x64_iu8(true, a0, true, b1, acc01, false, false);
    acc10 = __builtin_amdgcn_wmma_i32_16x16x64_iu8(true, a1, true, b0, acc10, false, false);
    acc11 = __builtin_amdgcn_wmma_i32_16x16x64_iu8(true, a1, true, b1, acc11, false, false);
    __syncthreads();
    if (i + 2 < nk) ISSUE(i + 2);
  }
  #undef ISSUE

  const float wdq = wslot[1];
  const int rb0 = tMblk + wm*32, rb1 = rb0 + 16;
  #define STORE_TILE(ACC, RB, NB) do {                                        \
    if ((NB) < N){                                                            \
      int _n = (NB) + l16;                                                    \
      float _bb = bias ? bias[_n] : 0.f;                                      \
      _Pragma("unroll")                                                       \
      for (int j = 0; j < 8; ++j){                                            \
        int _row = (RB) + j + hi*8;                                           \
        Y[(size_t)_row*ldy + col0 + _n] = (float)(ACC)[j]*rowdq[_row]*wdq + _bb; \
      }                                                                       \
    }                                                                         \
  } while (0)
  STORE_TILE(acc00, rb0, tN);
  STORE_TILE(acc01, rb0, tN + 16);
  STORE_TILE(acc10, rb1, tN);
  STORE_TILE(acc11, rb1, tN + 16);
  #undef STORE_TILE
}

__global__ void k_ln_mod(const float* __restrict__ X, int Tin, int in_off,
                         float* __restrict__ Y, int Tout, int out_off,
                         const float* __restrict__ scale, const float* __restrict__ shift,
                         int estride, int T, int Hd){
  __shared__ float r1[256];
  __shared__ float r2[256];
  int b = blockIdx.x / T, t = blockIdx.x % T;
  const float* xr = X + ((size_t)(b*Tin + in_off + t))*Hd;
  float* yr       = Y + ((size_t)(b*Tout + out_off + t))*Hd;
  float s1 = 0.f, s2 = 0.f;
  for (int c = threadIdx.x; c < Hd; c += 256){ float v = xr[c]; s1 += v; s2 += v*v; }
  r1[threadIdx.x] = s1; r2[threadIdx.x] = s2; __syncthreads();
  for (int o = 128; o > 0; o >>= 1){
    if (threadIdx.x < o){ r1[threadIdx.x]+=r1[threadIdx.x+o]; r2[threadIdx.x]+=r2[threadIdx.x+o]; }
    __syncthreads();
  }
  float mu = r1[0]/(float)Hd;
  float rstd = rsqrtf(r2[0]/(float)Hd - mu*mu + 1e-6f);
  for (int c = threadIdx.x; c < Hd; c += 256){
    float xc = (xr[c]-mu)*rstd;
    float sh = shift ? shift[b*estride + c] : 0.f;
    yr[c] = xc*(1.f + scale[b*estride + c]) + sh;
  }
}

__global__ void k_resid(float* __restrict__ X, const float* __restrict__ Y, int Ty, int y_off,
                        const float* __restrict__ gate, int gstride, int T, int Hd, int total){
  int idx = blockIdx.x*blockDim.x + threadIdx.x;
  if (idx >= total) return;
  int c = idx % Hd, bt = idx / Hd, b = bt / T, t = bt % T;
  X[idx] += gate[b*gstride + c] * Y[((size_t)(b*Ty + y_off + t))*Hd + c];
}

__global__ void k_copyrows(const float* __restrict__ S, int Ts, int s_off,
                           float* __restrict__ D, int Td, int d_off,
                           int T, int Hd, int total){
  int idx = blockIdx.x*blockDim.x + threadIdx.x;
  if (idx >= total) return;
  int c = idx % Hd, bt = idx / Hd, b = bt / T, t = bt % T;
  D[((size_t)(b*Td + d_off + t))*Hd + c] = S[((size_t)(b*Ts + s_off + t))*Hd + c];
}

__global__ void k_scan(const float* __restrict__ F, const float* __restrict__ I,
                       float* __restrict__ Hout, int T, int C, int Bt){
  int tid = blockIdx.x*blockDim.x + threadIdx.x;
  if (tid >= Bt*C) return;
  int b = tid / C, c = tid % C;
  size_t base = (size_t)b*T*C + c;
  float h = 0.f;
  for (int t = 0; t < T; ++t){
    size_t idx = base + (size_t)t*C;
    float f  = sigm_(F[idx]);
    float iv = silu_(I[idx])*(1.f - f);
    h = f*h + iv;
    Hout[idx] = h;
  }
}

__global__ void k_gnorm(const float* __restrict__ Hs, const float* __restrict__ G,
                        const float* __restrict__ gw, float* __restrict__ O){
  __shared__ float red[64];
  int head = blockIdx.x % HEADS, row = blockIdx.x / HEADS;
  int c = head*64 + threadIdx.x;
  size_t idx = (size_t)row*HDIM + c;
  float v = Hs[idx];
  red[threadIdx.x] = v*v; __syncthreads();
  for (int o = 32; o > 0; o >>= 1){
    if (threadIdx.x < o) red[threadIdx.x] += red[threadIdx.x+o];
    __syncthreads();
  }
  float rs = rsqrtf(red[0]/64.f + 1e-8f);
  O[idx] = v*rs*gw[c]*silu_(G[idx]);
}

__global__ void k_silu_mul(const float* __restrict__ U, float* __restrict__ Z, int total){
  int idx = blockIdx.x*blockDim.x + threadIdx.x;
  if (idx >= total) return;
  int m = idx / INTER, j = idx % INTER;
  Z[idx] = silu_(U[(size_t)m*2*INTER + j]) * U[(size_t)m*2*INTER + INTER + j];
}

struct BitP  { const float *g, *w, *b; };
struct AttnP { BitP f, gp, i, o; const float* gnorm; };
struct MlpP  { BitP down, gate; };
struct DualP { const float *adac_b,*adac_w,*adax_b,*adax_w; AttnP attn; MlpP ff, ffc; };
struct SingleP { const float *ada_b,*ada_w; AttnP attn; MlpP mlp; BitP proj; };

struct Ctx {
  hipStream_t s;
  int8_t *qx, *qw;
  float *rowdq, *wpart, *wslot;
  float *bufA, *bufI, *bufF, *bufG, *bufH, *bufO, *bufAO, *bufU, *bufZ, *bufP;
};

static inline int cdiv(int a, int b){ return (a + b - 1)/b; }

static void bitlin(const Ctx& c, const float* X, int M, int K,
                   const float* g, const float* W, const float* b, int N,
                   float* Y, int ldy, int col0){
  size_t nw = (size_t)N*K;
  k_wq_partial<<<256,256,0,c.s>>>(W, nw, c.wpart);
  k_wq_fin<<<1,256,0,c.s>>>(c.wpart, nw, c.wslot);
  k_wq_quant<<<(int)((nw+255)/256),256,0,c.s>>>(W, c.wslot, c.qw, nw);
  k_rq<<<M,256,0,c.s>>>(X, g, c.qx, c.rowdq, K);
  dim3 gg(cdiv(N,128), M/64);
  k_gemm_i8<<<gg,256,0,c.s>>>(c.qx, c.qw, c.rowdq, c.wslot, b, Y, M, N, K, ldy, col0);
}

static void attn_run(const Ctx& c, const AttnP& a, const float* X, int T,
                     float* out, int ldo, int col0){
  int M = BATCH*T;
  bitlin(c, X, M, HDIM, a.i.g,  a.i.w,  nullptr, HDIM, c.bufI, HDIM, 0);
  bitlin(c, X, M, HDIM, a.f.g,  a.f.w,  nullptr, HDIM, c.bufF, HDIM, 0);
  bitlin(c, X, M, HDIM, a.gp.g, a.gp.w, nullptr, HDIM, c.bufG, HDIM, 0);
  k_scan<<<cdiv(BATCH*HDIM,256),256,0,c.s>>>(c.bufF, c.bufI, c.bufH, T, HDIM, BATCH);
  k_gnorm<<<M*HEADS,64,0,c.s>>>(c.bufH, c.bufG, a.gnorm, c.bufO);
  bitlin(c, c.bufO, M, HDIM, a.o.g, a.o.w, nullptr, HDIM, out, ldo, col0);
}

static void mlp_run(const Ctx& c, const MlpP& m, const float* X, int M,
                    float* out, int ldo, int col0){
  bitlin(c, X, M, HDIM, m.gate.g, m.gate.w, nullptr, 2*INTER, c.bufU, 2*INTER, 0);
  k_silu_mul<<<cdiv(M*INTER,256),256,0,c.s>>>(c.bufU, c.bufZ, M*INTER);
  bitlin(c, c.bufZ, M, INTER, m.down.g, m.down.w, nullptr, HDIM, out, ldo, col0);
}

extern "C" void kernel_launch(void* const* d_in, const int* in_sizes, int n_in,
                              void* d_out, int out_size, void* d_ws, size_t ws_size,
                              hipStream_t stream){
  (void)in_sizes; (void)n_in; (void)out_size; (void)ws_size;
  const float* x_in   = (const float*)d_in[0];
  const float* t_in   = (const float*)d_in[1];
  const float* ctx_in = (const float*)d_in[2];

  int ip = 3;
  auto NXT = [&]() -> const float* { return (const float*)d_in[ip++]; };
  auto getAttn = [&](AttnP& a){
    a.f.b=nullptr;  a.f.g=NXT();  a.f.w=NXT();
    a.gp.b=nullptr; a.gp.g=NXT(); a.gp.w=NXT();
    a.gnorm = NXT();
    a.i.b=nullptr;  a.i.g=NXT();  a.i.w=NXT();
    a.o.b=nullptr;  a.o.g=NXT();  a.o.w=NXT();
  };
  auto getMlp = [&](MlpP& m){
    m.down.b=nullptr; m.down.g=NXT(); m.down.w=NXT();
    m.gate.b=nullptr; m.gate.g=NXT(); m.gate.w=NXT();
  };
  DualP dual[3];
  for (int d = 0; d < 3; ++d){
    dual[d].adac_b=NXT(); dual[d].adac_w=NXT();
    dual[d].adax_b=NXT(); dual[d].adax_w=NXT();
    getAttn(dual[d].attn); getMlp(dual[d].ff); getMlp(dual[d].ffc);
  }
  const float* fada_b = NXT(); const float* fada_w = NXT();
  BitP flin; flin.b=NXT(); flin.g=NXT(); flin.w=NXT();
  SingleP sing[3];
  for (int s = 0; s < 3; ++s){
    sing[s].ada_b=NXT(); sing[s].ada_w=NXT();
    getAttn(sing[s].attn); getMlp(sing[s].mlp);
    sing[s].proj.b=NXT(); sing[s].proj.g=NXT(); sing[s].proj.w=NXT();
  }
  const float* tm1_b=NXT(); const float* tm1_w=NXT();
  const float* tm2_b=NXT(); const float* tm2_w=NXT();
  BitP xemb; xemb.b=NXT(); xemb.g=NXT(); xemb.w=NXT();

  char* base = (char*)d_ws;
  size_t off = 0;
  auto alloc = [&](size_t bytes) -> void* {
    void* p = base + off; off += (bytes + 255) & ~(size_t)255; return p;
  };
  float* emb  = (float*)alloc(BATCH*256*4);
  float* t1b  = (float*)alloc(BATCH*HDIM*4);
  float* temb = (float*)alloc(BATCH*HDIM*4);
  float* e1   = (float*)alloc(BATCH*6*HDIM*4);
  float* e2   = (float*)alloc(BATCH*6*HDIM*4);
  float* xt   = (float*)alloc((size_t)BATCH*TXX*HDIM*4);
  float* yt   = (float*)alloc((size_t)BATCH*TCC*HDIM*4);
  float* comb = (float*)alloc((size_t)BATCH*TT*HDIM*4);
  Ctx c; c.s = stream;
  c.bufA  = (float*)alloc((size_t)BATCH*TT*HDIM*4);
  c.bufI  = (float*)alloc((size_t)BATCH*TT*HDIM*4);
  c.bufF  = (float*)alloc((size_t)BATCH*TT*HDIM*4);
  c.bufG  = (float*)alloc((size_t)BATCH*TT*HDIM*4);
  c.bufH  = (float*)alloc((size_t)BATCH*TT*HDIM*4);
  c.bufO  = (float*)alloc((size_t)BATCH*TT*HDIM*4);
  c.bufAO = (float*)alloc((size_t)BATCH*TT*HDIM*4);
  c.bufU  = (float*)alloc((size_t)BATCH*TT*2*INTER*4);
  c.bufZ  = (float*)alloc((size_t)BATCH*TT*INTER*4);
  c.bufP  = (float*)alloc((size_t)BATCH*TT*2*HDIM*4);
  c.qx    = (int8_t*)alloc((size_t)BATCH*TT*INTER);
  c.qw    = (int8_t*)alloc((size_t)2*INTER*HDIM);
  c.rowdq = (float*)alloc(BATCH*TT*4);
  c.wpart = (float*)alloc(256*4);
  c.wslot = (float*)alloc(2*4);

  k_timestep<<<2,256,0,stream>>>(t_in, emb);
  k_linear<<<cdiv(BATCH*HDIM,256),256,0,stream>>>(emb, tm1_w, tm1_b, t1b, BATCH, HDIM, 256, 0);
  k_linear<<<cdiv(BATCH*HDIM,256),256,0,stream>>>(t1b, tm2_w, tm2_b, temb, BATCH, HDIM, HDIM, 1);

  bitlin(c, x_in, BATCH*TXX, 64, xemb.g, xemb.w, xemb.b, HDIM, xt, HDIM, 0);
  k_copyrows<<<cdiv(BATCH*TCC*HDIM,256),256,0,stream>>>(ctx_in, TCC,0, yt, TCC,0, TCC, HDIM, BATCH*TCC*HDIM);

  for (int d = 0; d < 3; ++d){
    const DualP& P = dual[d];
    k_linear<<<cdiv(BATCH*6*HDIM,256),256,0,stream>>>(temb, P.adax_w, P.adax_b, e1, BATCH, 6*HDIM, HDIM, 1);
    k_linear<<<cdiv(BATCH*6*HDIM,256),256,0,stream>>>(temb, P.adac_w, P.adac_b, e2, BATCH, 6*HDIM, HDIM, 1);
    k_ln_mod<<<BATCH*TCC,256,0,stream>>>(yt, TCC,0, c.bufA, TT,0,   e2,       e2+HDIM, 6*HDIM, TCC, HDIM);
    k_ln_mod<<<BATCH*TXX,256,0,stream>>>(xt, TXX,0, c.bufA, TT,TCC, e1,       e1+HDIM, 6*HDIM, TXX, HDIM);
    attn_run(c, P.attn, c.bufA, TT, c.bufAO, HDIM, 0);
    k_resid<<<cdiv(BATCH*TXX*HDIM,256),256,0,stream>>>(xt, c.bufAO, TT, TCC, e1+2*HDIM, 6*HDIM, TXX, HDIM, BATCH*TXX*HDIM);
    k_resid<<<cdiv(BATCH*TCC*HDIM,256),256,0,stream>>>(yt, c.bufAO, TT, 0,   e2+2*HDIM, 6*HDIM, TCC, HDIM, BATCH*TCC*HDIM);
    k_ln_mod<<<BATCH*TXX,256,0,stream>>>(xt, TXX,0, c.bufA, TXX,0, e1+3*HDIM, e1+4*HDIM, 6*HDIM, TXX, HDIM);
    mlp_run(c, P.ff, c.bufA, BATCH*TXX, c.bufI, HDIM, 0);
    k_resid<<<cdiv(BATCH*TXX*HDIM,256),256,0,stream>>>(xt, c.bufI, TXX, 0, e1+5*HDIM, 6*HDIM, TXX, HDIM, BATCH*TXX*HDIM);
    k_ln_mod<<<BATCH*TCC,256,0,stream>>>(yt, TCC,0, c.bufA, TCC,0, e2+3*HDIM, e2+4*HDIM, 6*HDIM, TCC, HDIM);
    mlp_run(c, P.ffc, c.bufA, BATCH*TCC, c.bufI, HDIM, 0);
    k_resid<<<cdiv(BATCH*TCC*HDIM,256),256,0,stream>>>(yt, c.bufI, TCC, 0, e2+5*HDIM, 6*HDIM, TCC, HDIM, BATCH*TCC*HDIM);
  }

  k_copyrows<<<cdiv(BATCH*TCC*HDIM,256),256,0,stream>>>(yt, TCC,0, comb, TT,0,   TCC, HDIM, BATCH*TCC*HDIM);
  k_copyrows<<<cdiv(BATCH*TXX*HDIM,256),256,0,stream>>>(xt, TXX,0, comb, TT,TCC, TXX, HDIM, BATCH*TXX*HDIM);

  for (int s = 0; s < 3; ++s){
    const SingleP& P = sing[s];
    k_linear<<<cdiv(BATCH*2*HDIM,256),256,0,stream>>>(temb, P.ada_w, P.ada_b, e1, BATCH, 2*HDIM, HDIM, 1);
    k_ln_mod<<<BATCH*TT,256,0,stream>>>(comb, TT,0, c.bufA, TT,0, e1, nullptr, 2*HDIM, TT, HDIM);
    attn_run(c, P.attn, c.bufA, TT, c.bufP, 2*HDIM, 0);
    mlp_run(c, P.mlp, c.bufA, BATCH*TT, c.bufP, 2*HDIM, HDIM);
    bitlin(c, c.bufP, BATCH*TT, 2*HDIM, P.proj.g, P.proj.w, P.proj.b, HDIM, c.bufAO, HDIM, 0);
    k_resid<<<cdiv(BATCH*TT*HDIM,256),256,0,stream>>>(comb, c.bufAO, TT, 0, e1+HDIM, 2*HDIM, TT, HDIM, BATCH*TT*HDIM);
  }

  k_linear<<<cdiv(BATCH*2*HDIM,256),256,0,stream>>>(temb, fada_w, fada_b, e1, BATCH, 2*HDIM, HDIM, 1);
  k_ln_mod<<<BATCH*TXX,256,0,stream>>>(comb, TT,TCC, c.bufA, TXX,0, e1+HDIM, e1, 2*HDIM, TXX, HDIM);
  bitlin(c, c.bufA, BATCH*TXX, HDIM, flin.g, flin.w, flin.b, 64, (float*)d_out, 64, 0);
}
